// EM_SupervisedContrastiveLoss_29411936043176
// MI455X (gfx1250) — matrix operand order repaired and checked by reading, not compiled, not run on hardware
//
#include <hip/hip_runtime.h>
#include <hip/hip_bf16.h>

#define B_ROWS 16384
#define NCLS   1000
#define NPAD   1024
#define FDIM   768
#define INV_T  10.0f
#define CE_W   1.0f
#define SDC_W  0.1f
#define MOM    0.9f
#define EPSV   1e-8f

typedef __attribute__((ext_vector_type(16))) __bf16 v16bf;
typedef __attribute__((ext_vector_type(8)))  float  v8f;

union BFrag { uint4 q[2]; v16bf v; };

// ---- workspace layout (bytes) ----
static constexpr size_t OFF_CBF16 = 0;                                   // 1024*768 bf16 (padded centers)
static constexpr size_t OFF_WSUM  = (size_t)NPAD * FDIM * 2;             // 1000 f32 (padded to 4096B)
static constexpr size_t OFF_RESP  = OFF_WSUM + 4096;                     // 16384 f32
static constexpr size_t OFF_LOSS  = OFF_RESP + (size_t)B_ROWS * 4;       // [0]=ce acc, [1]=sdc acc
static constexpr size_t OFF_ACC   = OFF_LOSS + 256;                      // 1000*768 f32 center accumulator
static constexpr size_t WS_END    = OFF_ACC + (size_t)NCLS * FDIM * 4;

// ---- 1) centers f32 -> bf16, zero-padded to 1024 classes ----
__global__ void k_convert_centers(const float* __restrict__ centers, __bf16* __restrict__ cb) {
    int idx = blockIdx.x * 256 + threadIdx.x;          // 0 .. 1024*768-1
    int c = idx / FDIM;
    float v = 0.0f;
    if (c < NCLS) v = centers[idx];
    cb[idx] = (__bf16)v;
}

// ---- 2) CE loss: one wave per row of outputs[16384][1000] ----
__global__ void k_ce(const float* __restrict__ outputs, const int* __restrict__ labels,
                     float* __restrict__ ce_acc) {
    const int wave = threadIdx.x >> 5, lane = threadIdx.x & 31;
    const int row = blockIdx.x * 8 + wave;
    const float4* o4 = (const float4*)(outputs + (size_t)row * NCLS);   // 1000 = 250 float4, 16B aligned
    float m = -3.4e38f;
    for (int i = lane; i < NCLS / 4; i += 32) {
        float4 v = o4[i];
        m = fmaxf(m, fmaxf(fmaxf(v.x, v.y), fmaxf(v.z, v.w)));
    }
    for (int off = 16; off >= 1; off >>= 1) m = fmaxf(m, __shfl_xor(m, off, 32));
    float z = 0.0f;
    for (int i = lane; i < NCLS / 4; i += 32) {
        float4 v = o4[i];
        z += __expf(v.x - m) + __expf(v.y - m) + __expf(v.z - m) + __expf(v.w - m);
    }
    for (int off = 16; off >= 1; off >>= 1) z += __shfl_xor(z, off, 32);
    if (lane == 0) {
        int lab = labels[row];
        float lp = outputs[(size_t)row * NCLS + lab] - m - __logf(z);
        unsafeAtomicAdd(ce_acc, -lp);
    }
}

// ---- 3) fused bf16 WMMA GEMM (features @ centers^T) + row softmax stats ----
__global__ void __launch_bounds__(256)
k_gemm_softmax(const float* __restrict__ features, const __bf16* __restrict__ cb,
               const int* __restrict__ labels, float* __restrict__ wsum,
               float* __restrict__ resp_out, float* __restrict__ sdc_acc) {
    __shared__ __align__(16) unsigned short sA[16 * FDIM];   // 16-row A tile, bf16 (24 KB)
    __shared__ __align__(16) float          sSf[16 * NPAD];  // f32 staging, then 16x1024 sims (64 KB)

    const int tid  = threadIdx.x;
    const int lane = tid & 31;
    const int wave = tid >> 5;
    const int l16  = lane & 15;
    const int half = lane >> 4;
    const int rowBase = blockIdx.x * 16;

    // --- stage A tile (f32, 48KB) global -> LDS via async copy engine (ASYNCcnt) ---
    {
        const float* src = features + (size_t)rowBase * FDIM;
        unsigned ldsBase = (unsigned)(unsigned long long)(void*)sSf;
        for (int i = tid; i < 16 * FDIM / 4; i += 256) {
            unsigned lds = ldsBase + (unsigned)i * 16u;
            unsigned long long g = (unsigned long long)(src + (size_t)i * 4);
            asm volatile("global_load_async_to_lds_b128 %0, %1, off"
                         :: "v"(lds), "v"(g) : "memory");
        }
        asm volatile("s_wait_asynccnt 0x0" ::: "memory");
    }
    __syncthreads();

    // --- convert staged f32 tile -> bf16 into sA ---
    for (int i = tid; i < 16 * FDIM / 4; i += 256) {
        float4 v = *(const float4*)&sSf[i * 4];
        int o = i * 4;
        __bf16 b0 = (__bf16)v.x, b1 = (__bf16)v.y, b2 = (__bf16)v.z, b3 = (__bf16)v.w;
        sA[o + 0] = __builtin_bit_cast(unsigned short, b0);
        sA[o + 1] = __builtin_bit_cast(unsigned short, b1);
        sA[o + 2] = __builtin_bit_cast(unsigned short, b2);
        sA[o + 3] = __builtin_bit_cast(unsigned short, b3);
    }
    __syncthreads();

    // --- 8 waves x 4 tile-pairs: A fragment reused across two N-tiles (2 WMMA per A load) ---
    for (int p = wave; p < NPAD / 32; p += 8) {
        const int nt0 = p * 2;
        v8f acc0 = {}, acc1 = {};
        // A lane layout (16x32 bf16): lanes<16 -> K {0..7,16..23}; lanes>=16 -> K {8..15,24..31}
        const unsigned short* aBase = &sA[l16 * FDIM + half * 8];
        // B lane layout (32x16 bf16): lanes<16 -> K {0..7,16..23}; lanes>=16 -> K {8..15,24..31}, N = lane%16
        const __bf16* b0 = cb + (size_t)(nt0 * 16 + l16) * FDIM + half * 8;
        const __bf16* b1 = b0 + (size_t)16 * FDIM;
#pragma unroll 4
        for (int kt = 0; kt < FDIM / 32; ++kt) {
            const int k0 = kt * 32;
            BFrag a, f0, f1;
            a.q[0]  = *(const uint4*)(aBase + k0);        // K = k0+half*8 .. +7
            a.q[1]  = *(const uint4*)(aBase + k0 + 16);   // K = k0+half*8+16 .. +23
            f0.q[0] = *(const uint4*)(b0 + k0);
            f0.q[1] = *(const uint4*)(b0 + k0 + 16);
            f1.q[0] = *(const uint4*)(b1 + k0);
            f1.q[1] = *(const uint4*)(b1 + k0 + 16);
            acc0 = __builtin_amdgcn_wmma_f32_16x16x32_bf16(
                false, a.v, false, f0.v, (short)0, acc0, false, false);
            acc1 = __builtin_amdgcn_wmma_f32_16x16x32_bf16(
                false, a.v, false, f1.v, (short)0, acc1, false, false);
        }
        // C layout: VGPR v -> row v + 8*half, col = lane%16
#pragma unroll
        for (int v = 0; v < 8; ++v) {
            int r = v + half * 8;
            sSf[r * NPAD + nt0 * 16 + l16]      = acc0[v];
            sSf[r * NPAD + nt0 * 16 + 16 + l16] = acc1[v];
        }
    }
    __syncthreads();

    // --- row softmax stats: 16 threads per row (contiguous lanes), mask padded cols >= 1000 ---
    const int rrow = tid >> 4;
    const int sub  = tid & 15;
    float m = -3.4e38f;
    for (int c = sub; c < NCLS; c += 16) m = fmaxf(m, sSf[rrow * NPAD + c]);
    for (int off = 8; off >= 1; off >>= 1) m = fmaxf(m, __shfl_xor(m, off, 16));
    float z = 0.0f;
    for (int c = sub; c < NCLS; c += 16) z += __expf((sSf[rrow * NPAD + c] - m) * INV_T);
    for (int off = 8; off >= 1; off >>= 1) z += __shfl_xor(z, off, 16);
    if (sub == 0) {
        const int grow = rowBase + rrow;
        const int lab  = labels[grow];
        float sl = sSf[rrow * NPAD + lab];
        float rp = __expf((sl - m) * INV_T) / z;          // resp[row, label]
        resp_out[grow] = rp;
        unsafeAtomicAdd(&wsum[lab], rp);
        unsafeAtomicAdd(sdc_acc, -__logf(rp + EPSV));
    }
}

// ---- 4) scatter: accC[label] += (1-mom) * resp/(wsum+eps) * features[row] ----
__global__ void k_scatter(const float* __restrict__ features, const int* __restrict__ labels,
                          const float* __restrict__ resp, const float* __restrict__ wsum,
                          float* __restrict__ accC) {
    const int wave = threadIdx.x >> 5, lane = threadIdx.x & 31;
    const int row = blockIdx.x * 8 + wave;
    const int lab = labels[row];
    const float coeff = resp[row] / (wsum[lab] + EPSV) * (1.0f - MOM);
    const float* fr = features + (size_t)row * FDIM;
    float* ar = accC + (size_t)lab * FDIM;
    for (int c = lane; c < FDIM; c += 32)
        unsafeAtomicAdd(&ar[c], coeff * fr[c]);
}

// ---- 5) finalize: momentum blend + total loss ----
__global__ void k_finalize(const float* __restrict__ centers, const float* __restrict__ accC,
                           const float* __restrict__ wsum, const float* __restrict__ loss_acc,
                           float* __restrict__ out) {
    const int idx = blockIdx.x * 256 + threadIdx.x;
    if (idx < NCLS * FDIM) {
        const int c = idx / FDIM;
        const float base = centers[idx];
        out[1 + idx] = (wsum[c] > 0.0f) ? (MOM * base + accC[idx]) : base;
    }
    if (idx == 0) {
        out[0] = CE_W * loss_acc[0] / (float)B_ROWS + SDC_W * loss_acc[1] / (float)B_ROWS;
    }
}

extern "C" void kernel_launch(void* const* d_in, const int* in_sizes, int n_in,
                              void* d_out, int out_size, void* d_ws, size_t ws_size,
                              hipStream_t stream) {
    const float* outputs  = (const float*)d_in[0];   // [16384,1000]
    const int*   labels   = (const int*)d_in[1];     // [16384]
    const float* features = (const float*)d_in[2];   // [16384,768]
    const float* centers  = (const float*)d_in[3];   // [1000,768]
    float* out = (float*)d_out;                      // [1 + 1000*768]
    char*  ws  = (char*)d_ws;

    __bf16* cb   = (__bf16*)(ws + OFF_CBF16);
    float*  wsum = (float*)(ws + OFF_WSUM);
    float*  resp = (float*)(ws + OFF_RESP);
    float*  loss = (float*)(ws + OFF_LOSS);
    float*  accC = (float*)(ws + OFF_ACC);

    hipMemsetAsync(ws + OFF_WSUM, 0, WS_END - OFF_WSUM, stream);
    k_convert_centers<<<(NPAD * FDIM) / 256, 256, 0, stream>>>(centers, cb);
    k_ce<<<B_ROWS / 8, 256, 0, stream>>>(outputs, labels, &loss[0]);
    k_gemm_softmax<<<B_ROWS / 16, 256, 0, stream>>>(features, cb, labels, wsum, resp, &loss[1]);
    k_scatter<<<B_ROWS / 8, 256, 0, stream>>>(features, labels, resp, wsum, accC);
    k_finalize<<<(NCLS * FDIM + 255) / 256, 256, 0, stream>>>(centers, accC, wsum, loss, out);
}
